// RTDETRTransformer_62010737819904
// MI455X (gfx1250) — compile-verified, matrix operand order checked by reading, not run on hardware
//
#include <hip/hip_runtime.h>
#include <hip/hip_bf16.h>
#include <math.h>

// ---------------------------------------------------------------------------
// Multi-scale deformable attention (RT-DETR) for MI455X / gfx1250.
// All heavy GEMMs run through v_wmma_f32_16x16x32_bf16 (fp32 -> bf16 on the
// fly, fp32 accumulate). Value projection is HBM-bound (~557MB @ 23.3TB/s).
// ---------------------------------------------------------------------------

typedef __attribute__((ext_vector_type(16))) __bf16        v16bf;
typedef __attribute__((ext_vector_type(8)))  float         v8f;
typedef __attribute__((ext_vector_type(4)))  float         mf4;
typedef __attribute__((ext_vector_type(4)))  unsigned int  mu4;
typedef __attribute__((ext_vector_type(4)))  unsigned short mus4;

#define BM 128
#define BN 128
#define BK 64
#define AS (BK + 8)   // LDS row stride in bf16 elems: 72*2 = 144B = 9*16B (aligned b128 frags)

__device__ __forceinline__ unsigned short f2bf(float f) {
    union { float f; unsigned int u; } c; c.f = f;
    unsigned int u = c.u + 0x7FFFu + ((c.u >> 16) & 1u);   // RNE
    return (unsigned short)(u >> 16);
}

// Load a 16x32 bf16 operand fragment (A row-major [M][K] or B as [N][K]).
// Lane l owns row (l&15); K halves: kb..kb+7 in halves 0..7, kb+16..kb+23 in
// halves 8..15, with kb += 8 for lanes 16..31 (ISA 7.12.2, 16-bit A layout).
__device__ __forceinline__ v16bf frag_ld(const unsigned short* p) {
    union { mu4 u[2]; v16bf v; } f;
    f.u[0] = *(const mu4*)(p);        // 8 bf16
    f.u[1] = *(const mu4*)(p + 16);   // 8 bf16
    return f.v;
}

// C[M,N] = A[M,K] @ B[N,K]^T + bias[N]      (K multiple of 64, N multiple of 128)
__global__ __launch_bounds__(256) void wmma_gemm_bias(
        const float* __restrict__ A, const float* __restrict__ B,
        const float* __restrict__ bias, float* __restrict__ C,
        int M, int N, int K)
{
    __shared__ unsigned short As[BM * AS];
    __shared__ unsigned short Bs[BN * AS];

    const int t     = threadIdx.x;
    const int lane  = t & 31;
    const int wave  = t >> 5;
    const int wm    = (wave >> 1) * 32;   // wave M offset within block tile
    const int wn    = (wave & 1) * 64;    // wave N offset within block tile
    const int mBase = blockIdx.x * BM;
    const int nBase = blockIdx.y * BN;

    v8f acc[2][4] = {};

    for (int k0 = 0; k0 < K; k0 += BK) {
        // --- stage A (128x64 f32 -> bf16 LDS), 8 float4 per thread ---
        #pragma unroll
        for (int i = 0; i < 8; i++) {
            int lin = t + i * 256;                 // 0..2047 float4 slots
            int row = lin >> 4;                    // /16 vec-per-row
            int c4  = (lin & 15) << 2;
            mf4 val = {0.f, 0.f, 0.f, 0.f};
            int gr = mBase + row;
            if (gr < M) val = *(const mf4*)(A + (size_t)gr * K + k0 + c4);
            mus4 o = { f2bf(val.x), f2bf(val.y), f2bf(val.z), f2bf(val.w) };
            *(mus4*)&As[row * AS + c4] = o;
        }
        // --- stage B (128x64 f32 -> bf16 LDS); N tiles always fully covered ---
        #pragma unroll
        for (int i = 0; i < 8; i++) {
            int lin = t + i * 256;
            int row = lin >> 4;
            int c4  = (lin & 15) << 2;
            mf4 val = *(const mf4*)(B + (size_t)(nBase + row) * K + k0 + c4);
            mus4 o = { f2bf(val.x), f2bf(val.y), f2bf(val.z), f2bf(val.w) };
            *(mus4*)&Bs[row * AS + c4] = o;
        }
        // prefetch next K-slab of A (global_prefetch_b8)
        if (k0 + BK < K) {
            int gr = mBase + (t >> 1);
            if (gr < M) __builtin_prefetch(A + (size_t)gr * K + k0 + BK, 0, 1);
        }
        __syncthreads();

        #pragma unroll
        for (int kk = 0; kk < BK; kk += 32) {
            const int kb = kk + ((lane >> 4) << 3);
            const int r  = lane & 15;
            v16bf af[2], bf[4];
            #pragma unroll
            for (int mt = 0; mt < 2; mt++)
                af[mt] = frag_ld(&As[(wm + mt * 16 + r) * AS + kb]);
            #pragma unroll
            for (int nt = 0; nt < 4; nt++)
                bf[nt] = frag_ld(&Bs[(wn + nt * 16 + r) * AS + kb]);
            #pragma unroll
            for (int mt = 0; mt < 2; mt++)
                #pragma unroll
                for (int nt = 0; nt < 4; nt++)
                    acc[mt][nt] = __builtin_amdgcn_wmma_f32_16x16x32_bf16(
                        false, af[mt], false, bf[nt], (short)0, acc[mt][nt],
                        false, false);
        }
        __syncthreads();
    }

    // C/D layout (ISA 7.12.2): vgpr r, lane l -> m = r + 8*(l>>4), n = l&15
    const int cn  = lane & 15;
    const int rhi = (lane >> 4) * 8;
    #pragma unroll
    for (int mt = 0; mt < 2; mt++) {
        #pragma unroll
        for (int nt = 0; nt < 4; nt++) {
            int gn = nBase + wn + nt * 16 + cn;
            float bv = bias[gn];
            #pragma unroll
            for (int r = 0; r < 8; r++) {
                int gm = mBase + wm + mt * 16 + rhi + r;
                if (gm < M) C[(size_t)gm * N + gn] = acc[mt][nt][r] + bv;
            }
        }
    }
}

// ---------------------------------------------------------------------------
// Softmax + bilinear sampling + weighted sum.  One wave per (b, q, head);
// lane = head channel (hd = 32 = wave32) -> every corner gather is one
// coalesced 128B line from the projected value tensor.
// ---------------------------------------------------------------------------
__global__ __launch_bounds__(256) void msda_sample(
        const float* __restrict__ ref,   // [8,300,4,2]
        const float* __restrict__ off,   // [2400,256]  = (h*4+l)*4+p, xy
        const float* __restrict__ aw,    // [2400,128]  logits
        const float* __restrict__ v,     // [8,34000,256] (col = h*32+d)
        float* __restrict__ out)         // [2400,256]
{
    const int Hs[4]     = {160, 80, 40, 20};
    const int starts[4] = {0, 25600, 32000, 33600};

    const int wave = (blockIdx.x << 3) + (threadIdx.x >> 5);   // 0..19199
    const int lane = threadIdx.x & 31;
    const int h  = wave & 7;
    const int bq = wave >> 3;
    const int b  = bq / 300;

    // softmax over the 16 (level,point) logits of this head
    const float* lg = aw + bq * 128 + h * 16;
    float wgt[16];
    float mx = lg[0];
    #pragma unroll
    for (int j = 1; j < 16; j++) mx = fmaxf(mx, lg[j]);
    float s = 0.f;
    #pragma unroll
    for (int j = 0; j < 16; j++) { wgt[j] = __expf(lg[j] - mx); s += wgt[j]; }
    const float inv = 1.0f / s;

    const float* vb  = v + (size_t)b * 34000 * 256 + h * 32 + lane;
    const float* ofb = off + bq * 256 + h * 32;

    float acc = 0.f;
    #pragma unroll
    for (int l = 0; l < 4; l++) {
        const int   H = Hs[l], W = Hs[l], st = starts[l];
        const float rx = ref[(bq * 4 + l) * 2 + 0];
        const float ry = ref[(bq * 4 + l) * 2 + 1];
        #pragma unroll
        for (int p = 0; p < 4; p++) {
            const float ox = ofb[(l * 4 + p) * 2 + 0];
            const float oy = ofb[(l * 4 + p) * 2 + 1];
            const float x  = (rx + ox / (float)W) * (float)W - 0.5f;
            const float y  = (ry + oy / (float)H) * (float)H - 0.5f;
            const float xf = floorf(x), yf = floorf(y);
            const float wx = x - xf,    wy = y - yf;
            const int x0 = (int)xf, y0 = (int)yf;
            float g = 0.f;
            if ((unsigned)x0       < (unsigned)W && (unsigned)y0       < (unsigned)H)
                g += (1.f - wx) * (1.f - wy) * vb[(size_t)(st + y0 * W + x0) * 256];
            if ((unsigned)(x0 + 1) < (unsigned)W && (unsigned)y0       < (unsigned)H)
                g += wx * (1.f - wy)         * vb[(size_t)(st + y0 * W + x0 + 1) * 256];
            if ((unsigned)x0       < (unsigned)W && (unsigned)(y0 + 1) < (unsigned)H)
                g += (1.f - wx) * wy         * vb[(size_t)(st + (y0 + 1) * W + x0) * 256];
            if ((unsigned)(x0 + 1) < (unsigned)W && (unsigned)(y0 + 1) < (unsigned)H)
                g += wx * wy                 * vb[(size_t)(st + (y0 + 1) * W + x0 + 1) * 256];
            acc += wgt[l * 4 + p] * inv * g;
        }
    }
    out[bq * 256 + h * 32 + lane] = acc;
}

// ---------------------------------------------------------------------------
extern "C" void kernel_launch(void* const* d_in, const int* in_sizes, int n_in,
                              void* d_out, int out_size, void* d_ws, size_t ws_size,
                              hipStream_t stream) {
    const float* query = (const float*)d_in[0];   // [8,300,256]
    const float* refp  = (const float*)d_in[1];   // [8,300,4,2]
    const float* value = (const float*)d_in[2];   // [8,34000,256]
    const float* Wv    = (const float*)d_in[3];   // [256,256]
    const float* bv    = (const float*)d_in[4];
    const float* Woff  = (const float*)d_in[5];   // [256,256]
    const float* boff  = (const float*)d_in[6];
    const float* Wattn = (const float*)d_in[7];   // [128,256]
    const float* battn = (const float*)d_in[8];
    const float* Wout  = (const float*)d_in[9];   // [256,256]
    const float* bout  = (const float*)d_in[10];
    float* outp = (float*)d_out;                  // [8,300,256]

    // workspace layout (fp32): v proj | off | aw | attn_out  (~285 MB)
    float* ws    = (float*)d_ws;
    float* vproj = ws;                                   // 272000*256
    float* offp  = vproj + (size_t)272000 * 256;         // 2400*256
    float* awp   = offp  + (size_t)2400 * 256;           // 2400*128
    float* attn  = awp   + (size_t)2400 * 128;           // 2400*256

    dim3 blk(256);
    // 1. value projection: [272000,256] x [256,256]^T  (memory-bound, bf16 WMMA)
    wmma_gemm_bias<<<dim3(272000 / BM, 256 / BN), blk, 0, stream>>>(
        value, Wv, bv, vproj, 272000, 256, 256);
    // 2. sampling offsets: [2400,256] x [256,256]^T
    wmma_gemm_bias<<<dim3((2400 + BM - 1) / BM, 256 / BN), blk, 0, stream>>>(
        query, Woff, boff, offp, 2400, 256, 256);
    // 3. attention logits: [2400,256] x [128,256]^T
    wmma_gemm_bias<<<dim3((2400 + BM - 1) / BM, 128 / BN), blk, 0, stream>>>(
        query, Wattn, battn, awp, 2400, 128, 256);
    // 4. softmax + bilinear sample + weighted sum -> attn [2400,256]
    msda_sample<<<dim3(2400), blk, 0, stream>>>(refp, offp, awp, vproj, attn);
    // 5. output projection -> d_out
    wmma_gemm_bias<<<dim3((2400 + BM - 1) / BM, 256 / BN), blk, 0, stream>>>(
        attn, Wout, bout, outp, 2400, 256, 256);
}
